// GAT_83614423319311
// MI455X (gfx1250) — matrix operand order repaired and checked by reading, compiled, not run on hardware
//
#include <hip/hip_runtime.h>
#include <hip/hip_bf16.h>
#include <math.h>

typedef __attribute__((ext_vector_type(16))) _Float16 v16h;
typedef __attribute__((ext_vector_type(8)))  float    v8f;
typedef __attribute__((ext_vector_type(4)))  unsigned v4u;
typedef __attribute__((ext_vector_type(8)))  int      v8i_;
typedef __attribute__((ext_vector_type(4)))  int      v4i_;

#if defined(__has_builtin)
#if __has_builtin(__builtin_amdgcn_tensor_load_to_lds)
#define HAVE_TDM 1
#endif
#endif
#ifndef HAVE_TDM
#define HAVE_TDM 0
#endif

// Problem constants (match reference)
constexpr int N_ROI = 116;
constexpr int Gnum  = 128;
constexpr int DEG   = 32;
constexpr int Nn    = Gnum * N_ROI;          // 14848
constexpr int En    = Gnum * N_ROI * DEG;    // 475136
constexpr int HID   = 64;
constexpr int HEADS = 4;
constexpr int HC    = 256;
constexpr int EDIM  = 5;
constexpr int EMB   = 16;
constexpr int KIN   = N_ROI + EMB;           // 132

__device__ __forceinline__ float leaky01(float v) { return v >= 0.f ? v : 0.01f * v; }
__device__ __forceinline__ float leaky20(float v) { return v >= 0.f ? v : 0.20f * v; }

// ---------------------------------------------------------------------------
// fill kernel (init workspace regions; d_ws is poisoned by harness)
// ---------------------------------------------------------------------------
__global__ void fill_u32(unsigned* p, unsigned n, unsigned v) {
  unsigned i = blockIdx.x * 256u + threadIdx.x;
  if (i < n) p[i] = v;
}

// ---------------------------------------------------------------------------
// TDM: DMA a 16-row x 32-col f32 tile (row stride = K elements) into LDS.
// D# per CDNA5 ISA ch.8: group0 {count=1, lds_addr, global_addr, type=2},
// group1 {data_size=4B, tensor 32x16, tile 32x16, dim0_stride=K}, groups
// 2/3 zero (2-D tensor).  Issued once per wave (EXEC ignored by TDM).
// This toolchain exposes the 6-arg builtin (extra int32x8 group before cpol).
// ---------------------------------------------------------------------------
#if HAVE_TDM
__device__ __forceinline__ void tdm_load_tile(const float* gptr, unsigned lds_off,
                                              int K)
{
  const unsigned long long ga = (unsigned long long)(uintptr_t)gptr;
  v4u g0;
  g0[0] = 1u;                                        // count=1 (valid user D#)
  g0[1] = lds_off;                                   // lds_addr (bytes)
  g0[2] = (unsigned)(ga & 0xffffffffu);              // global_addr[31:0]
  g0[3] = (unsigned)((ga >> 32) & 0x01ffffffu)       // global_addr[56:32]
          | 0x80000000u;                             // type=2 ("image")
  v8i_ g1;
  g1[0] = (int)(2u << 16);                           // data_size=2 -> 4 bytes
  g1[1] = (int)(32u << 16);                          // tensor_dim0 = 32
  g1[2] = (int)(16u << 16);                          // tensor_dim1 = 16
  g1[3] = (int)(32u << 16);                          // tile_dim0   = 32
  g1[4] = 16;                                        // tile_dim1   = 16
  g1[5] = K;                                         // tensor_dim0_stride = K
  g1[6] = 0;
  g1[7] = 0;
  const v4i_ z4 = {0, 0, 0, 0};
  const v8i_ z8 = {0, 0, 0, 0, 0, 0, 0, 0};
  __builtin_amdgcn_tensor_load_to_lds(g0, g1, z4, z4, z8, 0);
}
#endif

// ---------------------------------------------------------------------------
// Embed: h0[N,64] = leaky( concat(x, group_emb[node_group]) @ W_embed + b )
// grid.x = N/16, block = 128 (4 waves, each wave owns a 16-col tile of 64)
// K = 132 padded to 160, staged f32->f16 through LDS (gather/concat A, so
// no TDM here).
// ---------------------------------------------------------------------------
__global__ void embed_kernel(const float* __restrict__ x,
                             const int*   __restrict__ node_group,
                             const float* __restrict__ gemb,
                             const float* __restrict__ W,
                             const float* __restrict__ bias,
                             float* __restrict__ h0)
{
  __shared__ _Float16 sA[16 * 32];
  const int tid = threadIdx.x, lane = tid & 31, wv = tid >> 5;
  const size_t row0 = (size_t)blockIdx.x * 16;
  const int colbase = wv * 16;
  v8f acc = {};
  for (int k0 = 0; k0 < 160; k0 += 32) {
    for (int i = tid; i < 512; i += 128) {
      const int r = i >> 5;
      const int kk = (i & 31) + k0;
      const size_t row = row0 + r;
      float v = 0.f;
      if (kk < N_ROI)      v = x[row * N_ROI + kk];
      else if (kk < KIN)   v = gemb[(size_t)node_group[row] * EMB + (kk - N_ROI)];
      sA[i] = (_Float16)v;
    }
    __syncthreads();
    v16h a, b;
    const int m = lane & 15, kb = (lane >> 4) * 8;
#pragma unroll
    for (int i = 0; i < 16; ++i) {
      const int kk = (i < 8) ? (kb + i) : (16 + kb + (i - 8));
      a[i] = sA[m * 32 + kk];
    }
    const int nn = colbase + (lane & 15);
#pragma unroll
    for (int i = 0; i < 16; ++i) {
      const int kr = k0 + ((i < 8) ? (kb + i) : (16 + kb + (i - 8)));
      float v = (kr < KIN) ? W[(size_t)kr * HID + nn] : 0.f;
      b[i] = (_Float16)v;
    }
    acc = __builtin_amdgcn_wmma_f32_16x16x32_f16(false, a, false, b, (short)0, acc,
                                                 false, false);
    __syncthreads();
  }
  const int n = colbase + (lane & 15);
#pragma unroll
  for (int r = 0; r < 8; ++r) {
    const int m = r + ((lane >> 4) << 3);
    h0[(row0 + m) * HID + n] = leaky01(acc[r] + bias[n]);
  }
}

// ---------------------------------------------------------------------------
// GINE edge message, fused edge-encoder: K=5 projection computed on the fly
// (never materialize eemb[E,64]).  64 threads per edge, block = 4 edges.
// ---------------------------------------------------------------------------
__global__ void gine_agg(const float* __restrict__ h0,
                         const float* __restrict__ ea,
                         const float* __restrict__ We,
                         const float* __restrict__ be,
                         const int*   __restrict__ ei,
                         float* __restrict__ agg)
{
  const int tid = threadIdx.x;
  const size_t e = (size_t)blockIdx.x * 4 + (tid >> 6);
  const int c = tid & 63;
  const int src = ei[e], dst = ei[(size_t)En + e];
  float ev = be[c];
#pragma unroll
  for (int d = 0; d < EDIM; ++d) ev += ea[e * EDIM + d] * We[d * HID + c];
  ev = leaky01(ev);
  float msg = h0[(size_t)src * HID + c] + ev;
  msg = fmaxf(msg, 0.f);
  atomicAdd(&agg[(size_t)dst * HID + c], msg);
}

// ---------------------------------------------------------------------------
// GINE MLP + LayerNorm, fully fused per 16-row tile:
//   t  = leaky((h0+agg) @ W1 + b1)        (WMMA, via LDS)
//   h2 = leaky(t @ W2 + b2)               (WMMA, t chained through LDS)
//   out = LN(h2) * g + b
// grid.x = N/16, block = 128.
// ---------------------------------------------------------------------------
__global__ void gine_mlp_ln(const float* __restrict__ h0,
                            const float* __restrict__ agg,
                            const float* __restrict__ W1, const float* __restrict__ b1,
                            const float* __restrict__ W2, const float* __restrict__ b2,
                            const float* __restrict__ g,  const float* __restrict__ be,
                            float* __restrict__ out)
{
  __shared__ _Float16 sX[16 * 64];
  __shared__ _Float16 sT[16 * 64];
  __shared__ float    sH[16 * 64];
  __shared__ float    sMu[16], sRs[16];
  const int tid = threadIdx.x, lane = tid & 31, wv = tid >> 5;
  const size_t row0 = (size_t)blockIdx.x * 16;
  const int colbase = wv * 16;
  const int m = lane & 15, kb = (lane >> 4) * 8;

  for (int i = tid; i < 16 * 64; i += 128) {
    const size_t idx = (row0 + (i >> 6)) * HID + (i & 63);
    sX[i] = (_Float16)(h0[idx] + agg[idx]);
  }
  __syncthreads();

  v8f acc = {};
#pragma unroll
  for (int k0 = 0; k0 < 64; k0 += 32) {
    v16h a, b;
#pragma unroll
    for (int i = 0; i < 16; ++i) {
      const int kk = (i < 8) ? (kb + i) : (16 + kb + (i - 8));
      a[i] = sX[m * 64 + k0 + kk];
    }
    const int nn = colbase + (lane & 15);
#pragma unroll
    for (int i = 0; i < 16; ++i) b[i] = (_Float16)W1[(size_t)(k0 + ((i < 8) ? (kb + i) : (16 + kb + (i - 8)))) * HID + nn];
    acc = __builtin_amdgcn_wmma_f32_16x16x32_f16(false, a, false, b, (short)0, acc,
                                                 false, false);
  }
  {
    const int n = colbase + (lane & 15);
#pragma unroll
    for (int r = 0; r < 8; ++r) {
      const int mm = r + ((lane >> 4) << 3);
      sT[mm * 64 + n] = (_Float16)leaky01(acc[r] + b1[n]);
    }
  }
  __syncthreads();

  v8f acc2 = {};
#pragma unroll
  for (int k0 = 0; k0 < 64; k0 += 32) {
    v16h a, b;
#pragma unroll
    for (int i = 0; i < 16; ++i) {
      const int kk = (i < 8) ? (kb + i) : (16 + kb + (i - 8));
      a[i] = sT[m * 64 + k0 + kk];
    }
    const int nn = colbase + (lane & 15);
#pragma unroll
    for (int i = 0; i < 16; ++i) b[i] = (_Float16)W2[(size_t)(k0 + ((i < 8) ? (kb + i) : (16 + kb + (i - 8)))) * HID + nn];
    acc2 = __builtin_amdgcn_wmma_f32_16x16x32_f16(false, a, false, b, (short)0, acc2,
                                                  false, false);
  }
  {
    const int n = colbase + (lane & 15);
#pragma unroll
    for (int r = 0; r < 8; ++r) {
      const int mm = r + ((lane >> 4) << 3);
      sH[mm * 64 + n] = leaky01(acc2[r] + b2[n]);
    }
  }
  __syncthreads();

  if (tid < 16) {
    float mu = 0.f;
    for (int c = 0; c < 64; ++c) mu += sH[tid * 64 + c];
    mu *= (1.f / 64.f);
    float var = 0.f;
    for (int c = 0; c < 64; ++c) {
      const float d = sH[tid * 64 + c] - mu;
      var += d * d;
    }
    var *= (1.f / 64.f);
    sMu[tid] = mu;
    sRs[tid] = rsqrtf(var + 1e-5f);
  }
  __syncthreads();
  for (int i = tid; i < 16 * 64; i += 128) {
    const int r = i >> 6, c = i & 63;
    out[(row0 + r) * HID + c] = (sH[i] - sMu[r]) * sRs[r] * g[c] + be[c];
  }
}

// ---------------------------------------------------------------------------
// Generic WMMA GEMM: C[M,NC] = A[M,K] @ B[K,NC]  (f32 io, f16 MACs)
// grid = (M/16, NC/64), block = 128.
// A tiles DMA'd by the Tensor Data Mover into a double-buffered LDS tile,
// pipelined on TENSORcnt (wave 0 issues tile t+1, waits tensorcnt<=1 while
// tile t is consumed).  Falls back to cooperative loads if the TDM builtin
// is unavailable.
// ---------------------------------------------------------------------------
__global__ void wmma_gemm(const float* __restrict__ A, const float* __restrict__ B,
                          float* __restrict__ C, int K, int NC)
{
  __shared__ float sAf[2][16 * 32];
  const int tid = threadIdx.x, lane = tid & 31, wv = tid >> 5;
  const size_t row0 = (size_t)blockIdx.x * 16;
  const int colbase = blockIdx.y * 64 + wv * 16;
  const int m = lane & 15, kb = (lane >> 4) * 8;
  const int nk = K >> 5;
  v8f acc = {};

#if HAVE_TDM
  if (wv == 0) {
    tdm_load_tile(A + row0 * (size_t)K, (unsigned)(uintptr_t)&sAf[0][0], K);
  }
  for (int t = 0; t < nk; ++t) {
    if (wv == 0) {
      if (t + 1 < nk)
        tdm_load_tile(A + row0 * (size_t)K + (size_t)(t + 1) * 32,
                      (unsigned)(uintptr_t)&sAf[(t + 1) & 1][0], K);
      if (t + 1 < nk)
        __builtin_amdgcn_s_wait_tensorcnt(1);  // in-order: tile t complete
      else
        __builtin_amdgcn_s_wait_tensorcnt(0);
    }
    __syncthreads();
    const float* sa = sAf[t & 1];
    v16h a, b;
#pragma unroll
    for (int i = 0; i < 16; ++i) {
      const int kk = (i < 8) ? (kb + i) : (16 + kb + (i - 8));
      a[i] = (_Float16)sa[m * 32 + kk];
    }
    const int kr = t * 32 + (lane & 15) + ((lane >> 4) << 4);
    const float* __restrict__ brow = B + (size_t)(t * 32) * NC + colbase + (lane & 15);
    __builtin_prefetch(B + (size_t)(kr + 32 < K ? kr + 32 : kr) * NC + colbase, 0, 1);
#pragma unroll
    for (int i = 0; i < 16; ++i) b[i] = (_Float16)brow[(size_t)((i < 8) ? (kb + i) : (16 + kb + (i - 8))) * NC];
    acc = __builtin_amdgcn_wmma_f32_16x16x32_f16(false, a, false, b, (short)0, acc,
                                                 false, false);
    __syncthreads();
  }
#else
  for (int t = 0; t < nk; ++t) {
    const int k0 = t * 32;
    for (int i = tid; i < 512; i += 128) {
      const int r = i >> 5, kk = (i & 31) + k0;
      sAf[0][i] = A[(row0 + r) * (size_t)K + kk];
    }
    __syncthreads();
    v16h a, b;
#pragma unroll
    for (int i = 0; i < 16; ++i) {
      const int kk = (i < 8) ? (kb + i) : (16 + kb + (i - 8));
      a[i] = (_Float16)sAf[0][m * 32 + kk];
    }
    const int kr = k0 + (lane & 15) + ((lane >> 4) << 4);
    const float* __restrict__ brow = B + (size_t)k0 * NC + colbase + (lane & 15);
#pragma unroll
    for (int i = 0; i < 16; ++i) b[i] = (_Float16)brow[(size_t)((i < 8) ? (kb + i) : (16 + kb + (i - 8))) * NC];
    acc = __builtin_amdgcn_wmma_f32_16x16x32_f16(false, a, false, b, (short)0, acc,
                                                 false, false);
    __syncthreads();
  }
#endif

  const int n = colbase + (lane & 15);
#pragma unroll
  for (int r = 0; r < 8; ++r) {
    const int mm = r + ((lane >> 4) << 3);
    C[(row0 + mm) * (size_t)NC + n] = acc[r];
  }
}

// ---------------------------------------------------------------------------
// GATv2 attention logits.  1 wave per edge (block = 8 edges).
// Edge projection e = edge_attr @ We (K=5) computed in registers.
// logit[e,h] = sum_c att[h,c] * leaky0.2( xl[src,h,c] + xr[dst,h,c] + e_hc )
// Also int-punned atomic-max of logit per (dst, head).
// ---------------------------------------------------------------------------
__global__ void attn_logits(const float* __restrict__ xl, const float* __restrict__ xr,
                            const float* __restrict__ ea, const float* __restrict__ We,
                            const float* __restrict__ att, const int* __restrict__ ei,
                            float* __restrict__ logits, unsigned* __restrict__ mmax)
{
  const int lane = threadIdx.x & 31;
  const size_t e = (size_t)blockIdx.x * 8 + (threadIdx.x >> 5);
  const int src = ei[e], dst = ei[(size_t)En + e];
  const int h = lane >> 3;
  const int base = lane * 8;  // 8 consecutive channels, same head per lane
  float a0 = ea[e * EDIM + 0], a1 = ea[e * EDIM + 1], a2 = ea[e * EDIM + 2],
        a3 = ea[e * EDIM + 3], a4 = ea[e * EDIM + 4];
  float part = 0.f;
#pragma unroll
  for (int j = 0; j < 8; ++j) {
    const int idx = base + j;
    const float ev = a0 * We[idx] + a1 * We[HC + idx] + a2 * We[2 * HC + idx] +
                     a3 * We[3 * HC + idx] + a4 * We[4 * HC + idx];
    float z = xl[(size_t)src * HC + idx] + xr[(size_t)dst * HC + idx] + ev;
    part += leaky20(z) * att[idx];
  }
  part += __shfl_xor(part, 1, 32);
  part += __shfl_xor(part, 2, 32);
  part += __shfl_xor(part, 4, 32);
  if ((lane & 7) == 0) {
    logits[e * HEADS + h] = part;
    const unsigned bits = __float_as_uint(part);
    const unsigned enc = (bits & 0x80000000u) ? ~bits : (bits | 0x80000000u);
    atomicMax(&mmax[(size_t)dst * HEADS + h], enc);
  }
}

// ---------------------------------------------------------------------------
// exp(logit - max) in place, atomicAdd per-(dst,head) denominator.
// ---------------------------------------------------------------------------
__global__ void attn_expsum(float* __restrict__ logits, const unsigned* __restrict__ mmax,
                            float* __restrict__ s, const int* __restrict__ ei)
{
  const size_t i = (size_t)blockIdx.x * 256 + threadIdx.x;  // over E*HEADS
  const size_t e = i >> 2;
  const int h = (int)(i & 3);
  const int dst = ei[(size_t)En + e];
  const unsigned enc = mmax[(size_t)dst * HEADS + h];
  const unsigned bits = (enc & 0x80000000u) ? (enc & 0x7fffffffu) : ~enc;
  float mv = __uint_as_float(bits);
  if (!(fabsf(mv) <= 3.0e38f)) mv = 0.f;  // mirrors where(isfinite(m), m, 0)
  const float ex = __expf(logits[i] - mv);
  logits[i] = ex;
  atomicAdd(&s[(size_t)dst * HEADS + h], ex);
}

// ---------------------------------------------------------------------------
// Weighted aggregation: acc[dst,:] += alpha * xl[src,:].  1 wave per edge.
// ---------------------------------------------------------------------------
__global__ void attn_agg(const float* __restrict__ exv, const float* __restrict__ s,
                         const float* __restrict__ xl, const int* __restrict__ ei,
                         float* __restrict__ acc)
{
  const int lane = threadIdx.x & 31;
  const size_t e = (size_t)blockIdx.x * 8 + (threadIdx.x >> 5);
  const int src = ei[e], dst = ei[(size_t)En + e];
  const int h = lane >> 3;
  const float alpha = exv[e * HEADS + h] / (s[(size_t)dst * HEADS + h] + 1e-16f);
  const int base = lane * 8;
#pragma unroll
  for (int j = 0; j < 8; ++j) {
    atomicAdd(&acc[(size_t)dst * HC + base + j],
              alpha * xl[(size_t)src * HC + base + j]);
  }
}

// ---------------------------------------------------------------------------
// BatchNorm pass 1: per-column sum / sumsq (bias folded in).  64 rows/block.
// ---------------------------------------------------------------------------
__global__ void bn_stats(const float* __restrict__ acc, const float* __restrict__ bias,
                         float* __restrict__ colsum, float* __restrict__ colsq)
{
  const int c = threadIdx.x;
  const size_t row0 = (size_t)blockIdx.x * 64;
  float s = 0.f, q = 0.f;
  for (int r = 0; r < 64; ++r) {
    const float v = acc[(row0 + r) * HC + c] + bias[c];
    s += v;
    q += v * v;
  }
  atomicAdd(&colsum[c], s);
  atomicAdd(&colsq[c], q);
}

// ---------------------------------------------------------------------------
// BatchNorm pass 2 + leaky.  grid = N, block = 256 (one row per block).
// ---------------------------------------------------------------------------
__global__ void bn_apply(const float* __restrict__ acc, const float* __restrict__ bias,
                         const float* __restrict__ colsum, const float* __restrict__ colsq,
                         const float* __restrict__ g, const float* __restrict__ be,
                         float* __restrict__ out)
{
  const int c = threadIdx.x;
  const size_t i = (size_t)blockIdx.x * HC + c;
  const float inv_n = 1.f / (float)Nn;
  const float mu = colsum[c] * inv_n;
  const float var = colsq[c] * inv_n - mu * mu;
  const float v = (acc[i] + bias[c] - mu) * rsqrtf(var + 1e-5f) * g[c] + be[c];
  out[i] = leaky01(v);
}

// ---------------------------------------------------------------------------
// Global mean pool over 116 nodes per graph + final FC [256 -> 2].
// grid = G, block = 256.
// ---------------------------------------------------------------------------
__global__ void pool_fc(const float* __restrict__ h, const float* __restrict__ W,
                        const float* __restrict__ b, float* __restrict__ out)
{
  __shared__ float p0[256], p1[256];
  const int g = blockIdx.x, tid = threadIdx.x;
  float s = 0.f;
  const size_t base = (size_t)g * N_ROI * HC + tid;
  for (int r = 0; r < N_ROI; ++r) s += h[base + (size_t)r * HC];
  s *= (1.0f / (float)N_ROI);
  p0[tid] = s * W[tid * 2 + 0];
  p1[tid] = s * W[tid * 2 + 1];
  __syncthreads();
  for (int off = 128; off > 0; off >>= 1) {
    if (tid < off) { p0[tid] += p0[tid + off]; p1[tid] += p1[tid + off]; }
    __syncthreads();
  }
  if (tid == 0) {
    out[g * 2 + 0] = p0[0] + b[0];
    out[g * 2 + 1] = p1[0] + b[1];
  }
}

// ---------------------------------------------------------------------------
// host side
// ---------------------------------------------------------------------------
static void run_fill(unsigned* p, size_t n, unsigned v, hipStream_t st) {
  fill_u32<<<(unsigned)((n + 255) / 256), 256, 0, st>>>(p, (unsigned)n, v);
}

extern "C" void kernel_launch(void* const* d_in, const int* in_sizes, int n_in,
                              void* d_out, int out_size, void* d_ws, size_t ws_size,
                              hipStream_t stream)
{
  const float* x          = (const float*)d_in[0];
  const int*   ei         = (const int*)d_in[1];
  const float* ea         = (const float*)d_in[2];
  // d_in[3] = batch (unused: graphs are contiguous 116-node blocks)
  const int*   node_group = (const int*)d_in[4];
  const float* group_emb  = (const float*)d_in[5];
  const float* W_embed    = (const float*)d_in[6];
  const float* b_embed    = (const float*)d_in[7];
  const float* We_enc     = (const float*)d_in[8];
  const float* be_enc     = (const float*)d_in[9];
  const float* W1         = (const float*)d_in[10];
  const float* b1         = (const float*)d_in[11];
  const float* W2         = (const float*)d_in[12];
  const float* b2         = (const float*)d_in[13];
  const float* ln_g       = (const float*)d_in[14];
  const float* ln_b       = (const float*)d_in[15];
  const float* lWl[2]   = { (const float*)d_in[16], (const float*)d_in[23] };
  const float* lWr[2]   = { (const float*)d_in[17], (const float*)d_in[24] };
  const float* lWe[2]   = { (const float*)d_in[18], (const float*)d_in[25] };
  const float* lAtt[2]  = { (const float*)d_in[19], (const float*)d_in[26] };
  const float* lB[2]    = { (const float*)d_in[20], (const float*)d_in[27] };
  const float* lBnG[2]  = { (const float*)d_in[21], (const float*)d_in[28] };
  const float* lBnB[2]  = { (const float*)d_in[22], (const float*)d_in[29] };
  const float* fc2_W      = (const float*)d_in[30];
  const float* fc2_b      = (const float*)d_in[31];

  float* ws = (float*)d_ws;
  size_t o = 0;
  float* h0   = ws + o; o += (size_t)Nn * HID;
  float* agg  = ws + o; o += (size_t)Nn * HID;
  float* h1   = ws + o; o += (size_t)Nn * HID;
  float* xl   = ws + o; o += (size_t)Nn * HC;
  float* xr   = ws + o; o += (size_t)Nn * HC;
  float* acc  = ws + o; o += (size_t)Nn * HC;
  float* hA   = ws + o; o += (size_t)Nn * HC;
  float* hB   = ws + o; o += (size_t)Nn * HC;
  float* exv  = ws + o; o += (size_t)En * HEADS;
  unsigned* mbuf = (unsigned*)(ws + o); o += (size_t)Nn * HEADS;
  float* sbuf = ws + o; o += (size_t)Nn * HEADS;
  float* colsum = ws + o; o += HC;   // colsum/colsq adjacent -> one fill
  float* colsq  = ws + o; o += HC;

  const unsigned ENC_NEG_INF = 0x007FFFFFu;  // ~bits(-inf)

  // ---- Embed ----
  embed_kernel<<<Nn / 16, 128, 0, stream>>>(x, node_group, group_emb, W_embed,
                                            b_embed, h0);
  // ---- GINE (fused edge-encoder) ----
  run_fill((unsigned*)agg, (size_t)Nn * HID, 0u, stream);
  gine_agg<<<En / 4, 256, 0, stream>>>(h0, ea, We_enc, be_enc, ei, agg);
  gine_mlp_ln<<<Nn / 16, 128, 0, stream>>>(h0, agg, W1, b1, W2, b2, ln_g, ln_b, h1);

  // ---- GATv2 layers ----
  const float* hin = h1;
  int K = HID;
  float* houts[2] = { hA, hB };
  for (int l = 0; l < 2; ++l) {
    dim3 gg(Nn / 16, HC / 64);
    wmma_gemm<<<gg, 128, 0, stream>>>(hin, lWl[l], xl, K, HC);
    wmma_gemm<<<gg, 128, 0, stream>>>(hin, lWr[l], xr, K, HC);
    run_fill(mbuf, (size_t)Nn * HEADS, ENC_NEG_INF, stream);
    run_fill((unsigned*)sbuf, (size_t)Nn * HEADS, 0u, stream);
    run_fill((unsigned*)acc, (size_t)Nn * HC, 0u, stream);
    attn_logits<<<En / 8, 256, 0, stream>>>(xl, xr, ea, lWe[l], lAtt[l], ei, exv, mbuf);
    attn_expsum<<<(En * HEADS) / 256, 256, 0, stream>>>(exv, mbuf, sbuf, ei);
    attn_agg<<<En / 8, 256, 0, stream>>>(exv, sbuf, xl, ei, acc);
    run_fill((unsigned*)colsum, 2 * HC, 0u, stream);
    bn_stats<<<Nn / 64, 256, 0, stream>>>(acc, lB[l], colsum, colsq);
    bn_apply<<<Nn, 256, 0, stream>>>(acc, lB[l], colsum, colsq, lBnG[l], lBnB[l],
                                     houts[l]);
    hin = houts[l];
    K = HC;
  }

  // ---- Pool + classifier ----
  pool_fc<<<Gnum, 256, 0, stream>>>(hB, fc2_W, fc2_b, (float*)d_out);
}
